// AR_RNN_GRU_74646531604735
// MI455X (gfx1250) — matrix operand order repaired and checked by reading, not compiled, not run on hardware
//
#include <hip/hip_runtime.h>
#include <hip/hip_bf16.h>

// ---------------------------------------------------------------------------
// AR GRU stack for MI455X (gfx1250, wave32, WMMA bf16 16x16x32)
//
// 3-layer GRU (U=768), batch 64, 64 warm + 64 AR steps. Weights ~21 MB bf16
// -> L2-resident; the problem is sequential over 128 steps, so per-step
// efficiency = L2->WGP bandwidth + pipeline latency.
//
// CDNA5 async data path: B fragments (shared by all 4 waves of a block) are
// staged into LDS with GLOBAL_LOAD_ASYNC_TO_LDS_B64, double-buffered and
// drained with s_wait_asynccnt, then consumed via ds_load_b128. This cuts
// redundant L2 reads 4x and overlaps the next K-chunk's weight DMA with the
// current chunk's WMMAs.
// ---------------------------------------------------------------------------

typedef __attribute__((ext_vector_type(16))) __bf16 bf16x16;
typedef __attribute__((ext_vector_type(8)))  float  f32x8;
typedef __attribute__((ext_vector_type(2)))  int    v2i_t;

typedef __attribute__((address_space(1))) v2i_t gas_v2i;   // global b64 payload
typedef __attribute__((address_space(3))) v2i_t las_v2i;   // LDS b64 payload

#if __has_builtin(__builtin_amdgcn_global_load_async_to_lds_b64)
#define HAVE_ASYNC_LDS 1
#else
#define HAVE_ASYNC_LDS 0
#endif

union FragB16 { bf16x16 v; uint4 q[2]; };

__device__ inline unsigned short f2bf_rne(float f) {
    union { float f; unsigned u; } c; c.f = f;
    unsigned u = c.u;
    unsigned r = u + 0x7FFFu + ((u >> 16) & 1u);   // round-to-nearest-even
    return (unsigned short)(r >> 16);
}

__device__ inline float sigmoidf_fast(float x) {
    return 1.0f / (1.0f + __expf(-x));
}

__device__ inline f32x8 wmma_bf16(bf16x16 a, bf16x16 b, f32x8 c) {
    return __builtin_amdgcn_wmma_f32_16x16x32_bf16(
        /*neg_a=*/false, a, /*neg_b=*/false, b,
        /*c_mod=*/(short)0, c, /*reuse_a=*/false, /*reuse_b=*/false);
}

template <int N>
__device__ inline void wait_asynccnt() {
#if __has_builtin(__builtin_amdgcn_s_wait_asynccnt)
    __builtin_amdgcn_s_wait_asynccnt(N);
#elif HAVE_ASYNC_LDS
    if constexpr (N == 0)      asm volatile("s_wait_asynccnt 0" ::: "memory");
    else if constexpr (N == 1) asm volatile("s_wait_asynccnt 1" ::: "memory");
    else                       asm volatile("s_wait_asynccnt 3" ::: "memory");
#endif
}

// Copy one 8-byte element into LDS: async DMA if available, else sync ld/st.
__device__ inline void copy_b64_to_lds(const unsigned long long* __restrict__ g,
                                       unsigned long long* __restrict__ l) {
#if HAVE_ASYNC_LDS
    __builtin_amdgcn_global_load_async_to_lds_b64(
        (gas_v2i*)g, (las_v2i*)l, /*offset=*/0, /*cpol=*/0);
#else
    *l = *g;
#endif
}

// Load a 16x32 bf16 A-fragment from row-major bf16 [rows, stride].
// WMMA A layout (16-bit, 16x32): M = lane%16 ; lane<16 holds K 0..7 & 16..23,
// lane>=16 holds K 8..15 & 24..31 -> two 16-byte contiguous loads per lane.
__device__ inline bf16x16 load_a_rowmajor(const unsigned short* __restrict__ base,
                                          long row_stride, int row, int k0, int hi8) {
    const unsigned short* p = base + (long)row * row_stride + k0 + hi8;
    FragB16 f;
    f.q[0] = *(const uint4*)(p);
    f.q[1] = *(const uint4*)(p + 16);
    return f.v;
}

// Read a staged 32x16 B-fragment back out of LDS (2x ds_load_b128 per lane).
__device__ inline bf16x16 load_b_lds(const unsigned long long* __restrict__ frag,
                                     int lane) {
    const uint4* p = (const uint4*)(frag + (long)lane * 4);
    FragB16 f;
    f.q[0] = p[0];
    f.q[1] = p[1];
    return f.v;
}

// ---------------------------------------------------------------------------
// One-time packing: W[K,N] fp32 (row-major) -> bf16 fragment-major image.
// Fragment (kc,nt) = 1KB contiguous: dword ((kc*NT+nt)*32 + lane)*8 + r.
// B layout (16-bit, 32x16): N = lane%16; lanes<16 dword r holds K 0..7 & 16..23,
// lanes>=16 dword r holds K 8..15 & 24..31 (pairs, matching the A layout).
// ---------------------------------------------------------------------------
__global__ void pack_weight_bf16(const float* __restrict__ W,
                                 unsigned* __restrict__ out, int K, int N) {
    const long NT = N >> 4, KC = K >> 5;
    const long total = KC * NT * 32 * 8;
    for (long idx = (long)blockIdx.x * blockDim.x + threadIdx.x; idx < total;
         idx += (long)gridDim.x * blockDim.x) {
        int  r    = (int)(idx & 7);
        int  lane = (int)((idx >> 3) & 31);
        long t    = idx >> 8;
        long nt   = t % NT;
        long kc   = t / NT;
        long n    = nt * 16 + (lane & 15);
        long k0   = kc * 32 + ((lane & 16) ? 8 : 0) + ((r & 4) ? 16 : 0) + 2 * (r & 3);
        unsigned short lo = f2bf_rne(W[k0 * N + n]);
        unsigned short hi = f2bf_rne(W[(k0 + 1) * N + n]);
        out[idx] = (unsigned)lo | ((unsigned)hi << 16);
    }
}

__global__ void f32_to_bf16_kernel(const float* __restrict__ in,
                                   unsigned short* __restrict__ out, long n) {
    for (long i = (long)blockIdx.x * blockDim.x + threadIdx.x; i < n;
         i += (long)gridDim.x * blockDim.x)
        out[i] = f2bf_rne(in[i]);
}

// Broadcast h0 [1,U] to [64,U] in both fp32 and bf16.
__global__ void init_state_kernel(const float* __restrict__ h0,
                                  float* __restrict__ hf,
                                  unsigned short* __restrict__ hb, int U) {
    long total = 64L * U;
    for (long i = (long)blockIdx.x * blockDim.x + threadIdx.x; i < total;
         i += (long)gridDim.x * blockDim.x) {
        float v = h0[i % U];
        hf[i] = v;
        hb[i] = f2bf_rne(v);
    }
}

// ---------------------------------------------------------------------------
// Fused GRU step for one layer.
// Grid: 48 blocks (one 16-wide unit-column strip each), 128 threads = 4 waves
// (one 16-row M-tile each). Per K-chunk, the block stages the three shared
// gate B-fragments (z,r,h = 3KB) into LDS (double-buffered, async), every
// wave WMMA-accumulates its six gate tiles, then gate math + h writeback.
//
// Async pipeline per chunk (uniform 3 async b64 per wave per chunk):
//   stage(kc+1) -> s_wait_asynccnt 3 (drains chunk kc, keeps kc+1 in flight)
//   -> barrier (publish) -> ds_load B + WMMA -> barrier (close WAR window)
// ---------------------------------------------------------------------------
__global__ __launch_bounds__(128)
void gru_step_kernel(const unsigned short* __restrict__ x_bf, long x_stride, int x_KC,
                     const unsigned short* __restrict__ h_bf,   // [64,768] bf16
                     const float* __restrict__ h_f,             // [64,768] fp32
                     const unsigned* __restrict__ WxP,          // packed [x_K, 2304]
                     const unsigned* __restrict__ WhP,          // packed [768, 2304]
                     const float* __restrict__ bi,
                     const float* __restrict__ br,
                     unsigned short* __restrict__ h_out_bf,
                     float* __restrict__ h_out_f) {
    const int  U  = 768;
    const long NT = 144;                 // 2304/16 column tiles
    const int  tid  = threadIdx.x;
    const int  lane = tid & 31;
    const int  wave = tid >> 5;          // M tile 0..3
    const long ntz  = blockIdx.x;        // z-gate column tile; r:+48, h:+96

    __shared__ unsigned long long sB[2][3 * 128];  // 2 x (3 fragments x 1KB)

    f32x8 acc[6];
    const f32x8 zero = {0.f, 0.f, 0.f, 0.f, 0.f, 0.f, 0.f, 0.f};
    #pragma unroll
    for (int i = 0; i < 6; ++i) acc[i] = zero;

    const int row = wave * 16 + (lane & 15);
    const int hi8 = (lane & 16) ? 8 : 0;

    // Stage the 3 gate fragments of K-chunk kc into LDS buffer b.
    // 384 b64 elements / 128 threads = exactly 3 per thread (3 per wave-lane),
    // so per-wave ASYNCcnt bookkeeping is uniform.
    auto stage = [&](const unsigned* __restrict__ packed, int kc, int b) {
        #pragma unroll
        for (int it = 0; it < 3; ++it) {
            int q = tid + it * 128;        // 0..383
            int f = q >> 7;                // fragment 0..2 (gate z/r/h)
            int d = q & 127;               // b64 index within fragment
            const unsigned* g =
                packed + (((long)kc * NT + ntz + 48 * f) << 8) + ((long)d << 1);
            copy_b64_to_lds((const unsigned long long*)g, &sB[b][q]);
        }
    };

    // One staged accumulation pass: acc[accBase + 2g] += A @ B_g over KC chunks.
    auto pass = [&](const unsigned* __restrict__ packed,
                    const unsigned short* __restrict__ Ab, long Astr,
                    int KC, int accBase) {
        stage(packed, 0, 0);
        for (int kc = 0; kc < KC; ++kc) {
            int cur = kc & 1;
            if (kc + 1 < KC) {
                stage(packed, kc + 1, cur ^ 1);
                wait_asynccnt<3>();
            } else {
                wait_asynccnt<0>();
            }
            __syncthreads();               // staged buffer visible to all waves
            bf16x16 a = load_a_rowmajor(Ab, Astr, row, kc * 32, hi8);
            #pragma unroll
            for (int g = 0; g < 3; ++g) {
                bf16x16 b = load_b_lds(&sB[cur][g * 128], lane);
                acc[accBase + 2 * g] = wmma_bf16(a, b, acc[accBase + 2 * g]);
            }
            __syncthreads();               // all reads done before re-staging
        }
    };

    // Input contribution: gx = x @ Wx   (acc 0,2,4 = z,r,h columns)
    pass(WxP, x_bf, x_stride, x_KC, 0);
    // Recurrent contribution: gh = h @ Wh  (acc 1,3,5)
    pass(WhP, h_bf, U, 24, 1);

    // Gate math. C/D layout: lane<16 -> M=j, lane>=16 -> M=j+8; N = lane%16.
    const int n = (int)ntz * 16 + (lane & 15);
    const float biz = bi[n], bir = bi[n + 768], bih = bi[n + 1536];
    const float brz = br[n], brr = br[n + 768], brh = br[n + 1536];
    const int rbase = wave * 16 + ((lane & 16) ? 8 : 0);

    #pragma unroll
    for (int j = 0; j < 8; ++j) {
        float z  = sigmoidf_fast(acc[0][j] + biz + acc[1][j] + brz);
        float rg = sigmoidf_fast(acc[2][j] + bir + acc[3][j] + brr);
        float hc = tanhf(acc[4][j] + bih + rg * (acc[5][j] + brh));
        long  off = (long)(rbase + j) * U + n;
        float ho  = h_f[off];
        float hn  = z * ho + (1.0f - z) * hc;
        h_out_f[off]  = hn;
        h_out_bf[off] = f2bf_rne(hn);
    }
}

// ---------------------------------------------------------------------------
// Dense readout + fused normalization of the prediction for the next AR step.
// Grid: 32 blocks (512/16 column tiles) x 128 threads (4 M-tile waves).
// Same async LDS staging, one 1KB fragment per chunk (1 async b64 per
// wave-lane per chunk -> s_wait_asynccnt 1).
// ---------------------------------------------------------------------------
__global__ __launch_bounds__(128)
void dense_step_kernel(const unsigned short* __restrict__ p_bf,  // [64,768]
                       const unsigned* __restrict__ WdP,         // packed [768,512]
                       const float* __restrict__ bd,
                       const float* __restrict__ mean,
                       const float* __restrict__ stdv,
                       float* __restrict__ out,                  // [64,64,512]
                       int t,
                       unsigned short* __restrict__ xnorm) {     // [64,512] bf16
    const int  U  = 768, D = 512, KC = 24;
    const long NT = 32;
    const int  tid  = threadIdx.x;
    const int  lane = tid & 31;
    const int  wave = tid >> 5;
    const long nt   = blockIdx.x;

    __shared__ unsigned long long sB[2][128];   // 2 x 1KB fragment

    f32x8 acc = {0.f, 0.f, 0.f, 0.f, 0.f, 0.f, 0.f, 0.f};
    const int row = wave * 16 + (lane & 15);
    const int hi8 = (lane & 16) ? 8 : 0;

    auto stage = [&](int kc, int b) {
        const unsigned* g = WdP + (((long)kc * NT + nt) << 8) + ((long)tid << 1);
        copy_b64_to_lds((const unsigned long long*)g, &sB[b][tid]);
    };

    stage(0, 0);
    for (int kc = 0; kc < KC; ++kc) {
        int cur = kc & 1;
        if (kc + 1 < KC) {
            stage(kc + 1, cur ^ 1);
            wait_asynccnt<1>();
        } else {
            wait_asynccnt<0>();
        }
        __syncthreads();
        bf16x16 a = load_a_rowmajor(p_bf, U, row, kc * 32, hi8);
        bf16x16 b = load_b_lds(&sB[cur][0], lane);
        acc = wmma_bf16(a, b, acc);
        __syncthreads();
    }

    const int n = (int)nt * 16 + (lane & 15);
    const float b  = bd[n];
    const float mu = mean[n];
    const float sd = stdv[n];
    const int rbase = wave * 16 + ((lane & 16) ? 8 : 0);

    #pragma unroll
    for (int j = 0; j < 8; ++j) {
        float pred = acc[j] + b;
        int   r    = rbase + j;
        out[(long)r * (64 * D) + (long)t * D + n] = pred;
        xnorm[(long)r * D + n] = f2bf_rne((pred - mu) / sd);
    }
}

// ---------------------------------------------------------------------------
// Host orchestration
// ---------------------------------------------------------------------------
extern "C" void kernel_launch(void* const* d_in, const int* in_sizes, int n_in,
                              void* d_out, int out_size, void* d_ws, size_t ws_size,
                              hipStream_t stream) {
    const int B = 64, T = 64, D = 512, U = 768, G = 3 * U;  // G = 2304

    const float* inputs = (const float*)d_in[0];
    const float* Wx[3]  = {(const float*)d_in[1], (const float*)d_in[6],  (const float*)d_in[11]};
    const float* Wh[3]  = {(const float*)d_in[2], (const float*)d_in[7],  (const float*)d_in[12]};
    const float* bi[3]  = {(const float*)d_in[3], (const float*)d_in[8],  (const float*)d_in[13]};
    const float* br[3]  = {(const float*)d_in[4], (const float*)d_in[9],  (const float*)d_in[14]};
    const float* h0[3]  = {(const float*)d_in[5], (const float*)d_in[10], (const float*)d_in[15]};
    const float* Wd   = (const float*)d_in[16];
    const float* bd   = (const float*)d_in[17];
    const float* mean = (const float*)d_in[18];
    const float* stdv = (const float*)d_in[19];
    float* out = (float*)d_out;

    // Workspace carve-out (256B aligned slices)
    char*  ws  = (char*)d_ws;
    size_t off = 0;
    auto carve = [&](size_t bytes) -> char* {
        char* p = ws + off;
        off = (off + bytes + 255) & ~(size_t)255;
        return p;
    };

    unsigned* WxP[3]; unsigned* WhP[3];
    WxP[0] = (unsigned*)carve((size_t)D * G * 2);             // K=512
    WhP[0] = (unsigned*)carve((size_t)U * G * 2);
    WxP[1] = (unsigned*)carve((size_t)U * G * 2);
    WhP[1] = (unsigned*)carve((size_t)U * G * 2);
    WxP[2] = (unsigned*)carve((size_t)U * G * 2);
    WhP[2] = (unsigned*)carve((size_t)U * G * 2);
    unsigned* WdP = (unsigned*)carve((size_t)U * D * 2);
    unsigned short* inputs_bf = (unsigned short*)carve((size_t)B * T * D * 2);
    unsigned short* h_bf[3][2];
    float*          h_f[3][2];
    for (int j = 0; j < 3; ++j)
        for (int p = 0; p < 2; ++p) {
            h_bf[j][p] = (unsigned short*)carve((size_t)B * U * 2);
            h_f[j][p]  = (float*)carve((size_t)B * U * 4);
        }
    unsigned short* xnorm = (unsigned short*)carve((size_t)B * D * 2);

    // --- one-time conversions ---
    auto pack = [&](const float* W, unsigned* dst, int K, int N) {
        long total  = (long)(K / 32) * (N / 16) * 32 * 8;
        int  blocks = (int)((total + 255) / 256);
        if (blocks > 2048) blocks = 2048;
        pack_weight_bf16<<<blocks, 256, 0, stream>>>(W, dst, K, N);
    };
    pack(Wx[0], WxP[0], D, G);
    pack(Wh[0], WhP[0], U, G);
    pack(Wx[1], WxP[1], U, G);
    pack(Wh[1], WhP[1], U, G);
    pack(Wx[2], WxP[2], U, G);
    pack(Wh[2], WhP[2], U, G);
    pack(Wd, WdP, U, D);

    f32_to_bf16_kernel<<<2048, 256, 0, stream>>>(inputs, inputs_bf, (long)B * T * D);

    int cur[3];
    for (int j = 0; j < 3; ++j) {
        init_state_kernel<<<192, 256, 0, stream>>>(h0[j], h_f[j][0], h_bf[j][0], U);
        cur[j] = 0;
    }

    const long in_row_stride = (long)T * D;  // row stride of inputs_bf [B,T,D]

    auto gru_layer = [&](int j, const unsigned short* x, long xs, int xKC) {
        int c = cur[j];
        gru_step_kernel<<<48, 128, 0, stream>>>(
            x, xs, xKC, h_bf[j][c], h_f[j][c], WxP[j], WhP[j], bi[j], br[j],
            h_bf[j][1 - c], h_f[j][1 - c]);
        cur[j] = 1 - c;
    };

    // --- warmup: 64 teacher-forced steps ---
    for (int t = 0; t < T; ++t) {
        gru_layer(0, inputs_bf + (long)t * D, in_row_stride, D / 32);
        gru_layer(1, h_bf[0][cur[0]], U, U / 32);
        gru_layer(2, h_bf[1][cur[1]], U, U / 32);
    }

    // pred0 + normalized input for the first AR step
    dense_step_kernel<<<32, 128, 0, stream>>>(h_bf[2][cur[2]], WdP, bd, mean, stdv,
                                              out, 0, xnorm);

    // --- autoregressive rollout: 63 more predictions ---
    for (int t = 1; t < 64; ++t) {
        gru_layer(0, xnorm, D, D / 32);
        gru_layer(1, h_bf[0][cur[0]], U, U / 32);
        gru_layer(2, h_bf[1][cur[1]], U, U / 32);
        dense_step_kernel<<<32, 128, 0, stream>>>(h_bf[2][cur[2]], WdP, bd, mean, stdv,
                                                  out, t, xnorm);
    }
}
